// Model_49692771615413
// MI455X (gfx1250) — matrix operand order repaired and checked by reading, not compiled, not run on hardware
//
#include <hip/hip_runtime.h>

typedef __attribute__((ext_vector_type(16))) _Float16 v16h;
typedef __attribute__((ext_vector_type(8)))  float    v8f;

#define MAX_FTS 30
#define ROWW    257   // 256 features + 1 psqt per table row
#define TILE    16    // positions per block (WMMA M)

__global__ __launch_bounds__(256)
void nnue_fused_kernel(const int*   __restrict__ wft,
                       const int*   __restrict__ bft,
                       const float* __restrict__ stm,
                       const float* __restrict__ ftw,
                       const float* __restrict__ ftb,
                       const float* __restrict__ fc1w,
                       const float* __restrict__ fc1b,
                       const float* __restrict__ fc2w,
                       const float* __restrict__ fc2b,
                       const float* __restrict__ fcow,
                       const float* __restrict__ fcob,
                       float*       __restrict__ out)
{
    // Padded rows (+8 halves = +16B) so row strides are not multiples of the
    // 256B LDS bank period -> conflict-free WMMA fragment loads.
    __shared__ _Float16 x_h [TILE][520];   // mixed+clamped inputs, f16
    __shared__ _Float16 w1_h[32][520];     // fc1_w as f16
    __shared__ _Float16 w2_h[32][40];      // fc2_w as f16
    __shared__ _Float16 h1_h[TILE][40];    // fc1 activations, f16
    __shared__ float    h2  [TILE][33];    // fc2 activations, f32
    __shared__ float    ps  [TILE];        // psqt term per position

    const int tid  = threadIdx.x;
    const int lane = tid & 31;
    const int wave = tid >> 5;

    // ---- Phase 0: stage fc1_w / fc2_w into LDS as f16 (cooperative) ----
    for (int t = tid; t < 32 * 512; t += 256)
        w1_h[t >> 9][t & 511] = (_Float16)fc1w[t];
    for (int t = tid; t < 32 * 32; t += 256)
        w2_h[t >> 5][t & 31] = (_Float16)fc2w[t];

    // ---- Phase 1: feature gather-accumulate (L2-resident table) ----
    // Each wave owns 2 positions; lane l holds columns l+32j, j=0..7.
    for (int q = 0; q < 2; ++q) {
        const int pl  = wave * 2 + q;            // local position 0..15
        const int pos = blockIdx.x * TILE + pl;  // global position

        float wacc[8], bacc[8];
        #pragma unroll
        for (int j = 0; j < 8; ++j) { wacc[j] = 0.f; bacc[j] = 0.f; }
        float wp = 0.f, bp = 0.f;                // psqt accumulators (uniform)

        const int* wi = wft + pos * MAX_FTS;
        const int* bi = bft + pos * MAX_FTS;
        for (int i = 0; i < MAX_FTS; ++i) {
            const int iw = wi[i];
            if (iw >= 0) {
                const float* r = ftw + (size_t)iw * ROWW;
                #pragma unroll
                for (int j = 0; j < 8; ++j) wacc[j] += r[lane + 32 * j];
                wp += r[256];
            }
            const int ib = bi[i];
            if (ib >= 0) {
                const float* r = ftw + (size_t)ib * ROWW;
                #pragma unroll
                for (int j = 0; j < 8; ++j) bacc[j] += r[lane + 32 * j];
                bp += r[256];
            }
        }

        const float s = stm[pos];
        if (lane == 0) ps[pl] = (wp - bp) * (0.5f - s);

        // STM mix + feature bias + clamp [0,1]; convert to f16 for WMMA.
        #pragma unroll
        for (int j = 0; j < 8; ++j) {
            const int   c    = lane + 32 * j;
            const float bias = ftb[c];
            const float xa = (1.f - s) * wacc[j] + s * bacc[j] + bias;
            const float xb = (1.f - s) * bacc[j] + s * wacc[j] + bias;
            x_h[pl][c]       = (_Float16)fminf(fmaxf(xa, 0.f), 1.f);
            x_h[pl][256 + c] = (_Float16)fminf(fmaxf(xb, 0.f), 1.f);
        }
    }
    __syncthreads();

    // ---- Phase 2: fc1 (16x512 @ 512x32) via v_wmma_f32_16x16x32_f16 ----
    // Waves 0 and 1 each own one 16-wide N-tile; EXEC all-ones per wave.
    if (tid < 64) {
        const int n0 = wave * 16;
        const int ar = lane & 15;                 // A row (= M)
        const int ak = (lane < 16) ? 0 : 8;       // A K-offset per ISA layout
        const int bn = n0 + (lane & 15);          // B column (= N)
        const int bk = (lane < 16) ? 0 : 8;       // B K-offset per ISA layout
        v8f c = {};
        #pragma unroll
        for (int kb = 0; kb < 16; ++kb) {
            const int k = kb * 32;
            v16h a, b;
            const _Float16* xp = &x_h[ar][k + ak];
            #pragma unroll
            for (int i = 0; i < 8; ++i) { a[i] = xp[i]; a[8 + i] = xp[16 + i]; }
            const _Float16* wq = &w1_h[bn][k + bk];
            #pragma unroll
            for (int i = 0; i < 8; ++i) { b[i] = wq[i]; b[8 + i] = wq[16 + i]; }
            c = __builtin_amdgcn_wmma_f32_16x16x32_f16(
                    false, a, false, b, (short)0, c, false, false);
        }
        // C layout: VGPR r -> M = r (lanes 0-15) / r+8 (lanes 16-31), N = lane&15
        const float bias = fc1b[bn];
        const int   mb   = (lane < 16) ? 0 : 8;
        #pragma unroll
        for (int r = 0; r < 8; ++r)
            h1_h[mb + r][bn] = (_Float16)fminf(fmaxf(c[r] + bias, 0.f), 1.f);
    }
    __syncthreads();

    // ---- Phase 3: fc2 (16x32 @ 32x32) — one WMMA per N-tile ----
    if (tid < 64) {
        const int n0 = wave * 16;
        const int ar = lane & 15;
        const int ak = (lane < 16) ? 0 : 8;
        const int bn = n0 + (lane & 15);
        const int bk = (lane < 16) ? 0 : 8;
        v16h a, b;
        const _Float16* xp = &h1_h[ar][ak];
        #pragma unroll
        for (int i = 0; i < 8; ++i) { a[i] = xp[i]; a[8 + i] = xp[16 + i]; }
        const _Float16* wq = &w2_h[bn][bk];
        #pragma unroll
        for (int i = 0; i < 8; ++i) { b[i] = wq[i]; b[8 + i] = wq[16 + i]; }
        v8f c = {};
        c = __builtin_amdgcn_wmma_f32_16x16x32_f16(
                false, a, false, b, (short)0, c, false, false);
        const float bias = fc2b[bn];
        const int   mb   = (lane < 16) ? 0 : 8;
        #pragma unroll
        for (int r = 0; r < 8; ++r)
            h2[mb + r][bn] = fminf(fmaxf(c[r] + bias, 0.f), 1.f);
    }
    __syncthreads();

    // ---- Phase 4: fco (32 -> 1) + psqt correction, store ----
    if (tid < TILE) {
        float acc = fcob[0];
        #pragma unroll
        for (int n = 0; n < 32; ++n) acc += h2[tid][n] * fcow[n];
        out[blockIdx.x * TILE + tid] = acc + ps[tid];
    }
}

extern "C" void kernel_launch(void* const* d_in, const int* in_sizes, int n_in,
                              void* d_out, int out_size, void* d_ws, size_t ws_size,
                              hipStream_t stream) {
    const int*   wft  = (const int*)  d_in[0];
    const int*   bft  = (const int*)  d_in[1];
    const float* stm  = (const float*)d_in[2];
    const float* ftw  = (const float*)d_in[3];
    const float* ftb  = (const float*)d_in[4];
    const float* fc1w = (const float*)d_in[5];
    const float* fc1b = (const float*)d_in[6];
    const float* fc2w = (const float*)d_in[7];
    const float* fc2b = (const float*)d_in[8];
    const float* fcow = (const float*)d_in[9];
    const float* fcob = (const float*)d_in[10];
    float* out = (float*)d_out;

    const int B = in_sizes[2];          // stm has one element per position
    const int grid = B / TILE;          // B = 16384 -> 1024 blocks

    nnue_fused_kernel<<<grid, 256, 0, stream>>>(
        wft, bft, stm, ftw, ftb, fc1w, fc1b, fc2w, fc2b, fcow, fcob, out);
}
